// ExpertRouter_77094662963654
// MI455X (gfx1250) — compile-verified, matrix operand order checked by reading, not run on hardware
//
#include <hip/hip_runtime.h>

#define N_TOKENS 8192
#define IN_DIM   1024
#define EX_DIM   4096
#define NEXP     8
#define TM       64                 // tokens per workgroup
#define MT       (TM / 16)          // 4 M-tiles
#define CHUNK    128                // hidden-dim chunk
#define NCHUNK   (EX_DIM / CHUNK)   // 32
#define KSTEPS1  (IN_DIM / 32)      // 32 WMMA k-steps for GEMM1
#define KSTEPS2  (CHUNK / 32)       // 4  WMMA k-steps per chunk for GEMM2

// Workspace layout (bf16, pre-swizzled into WMMA B-tile layout; offsets in ushorts)
#define W1S_OFF  ((size_t)0)
#define W1S_ELEMS ((size_t)NEXP * IN_DIM * EX_DIM)
#define W2S_OFF  (W1S_OFF + W1S_ELEMS)
#define W2S_ELEMS ((size_t)NEXP * EX_DIM * IN_DIM)
#define WG1S_OFF (W2S_OFF + W2S_ELEMS)
#define WG1S_ELEMS ((size_t)IN_DIM * EX_DIM)
// total ws: 142,606,336 bytes

typedef __attribute__((ext_vector_type(16))) __bf16       v16bf;
typedef __attribute__((ext_vector_type(8)))  float        v8f;
typedef __attribute__((ext_vector_type(4)))  unsigned int v4u;

static __device__ __forceinline__ unsigned short f32_bf16_bits(float f) {
  unsigned int u = __builtin_bit_cast(unsigned int, f);
  u += 0x7FFFu + ((u >> 16) & 1u);          // round-to-nearest-even
  return (unsigned short)(u >> 16);
}

// Within-B-tile (32x16) position of source element (kk=0..31, nn=0..15)
static __device__ __forceinline__ int b_tile_idx(int kk, int nn) {
  int lane = nn + (((kk >> 3) & 1) << 4);
  int e    = (kk & 7) | ((kk >> 4) << 3);
  return lane * 16 + e;
}

// A-tile (16x32) lane-striped layout (ISA 7.12.2)
static __device__ __forceinline__ int a_idx(int m, int kk) {
  int g    = kk >> 3;
  int lane = m + ((g & 1) << 4);
  int e    = ((g >> 1) << 3) + (kk & 7);
  return lane * 16 + e;
}

// Load this lane's 32B slot (16 bf16) -> two b128 loads (LDS or global)
static __device__ __forceinline__ v16bf load_slot(const unsigned short* slot) {
  union { v16bf bf; v4u u[2]; } ua;
  const v4u* p = (const v4u*)slot;
  ua.u[0] = p[0];
  ua.u[1] = p[1];
  return ua.bf;
}

// ---------------- Weight pre-convert/swizzle kernels (memory-bound pre-pass) -------------

// W1[e][k(1024)][n(4096)] -> tiles [e][ntile(256)][kstep(32)] of 512 ushorts
__global__ __launch_bounds__(256) void cvt_w1(const float* __restrict__ src,
                                              unsigned short* __restrict__ dst, size_t total) {
  for (size_t idx = (size_t)blockIdx.x * 256 + threadIdx.x; idx < total;
       idx += (size_t)gridDim.x * 256) {
    int n  = (int)(idx & (EX_DIM - 1));
    int k  = (int)((idx >> 12) & (IN_DIM - 1));
    int ee = (int)(idx >> 22);
    size_t tile = ((size_t)ee * 256 + (n >> 4)) * 32 + (k >> 5);
    dst[tile * 512 + b_tile_idx(k & 31, n & 15)] = f32_bf16_bits(src[idx]);
  }
}

// W2[e][k(4096)][n(1024)] -> tiles [e][kstep(128)][ntile(64)]
__global__ __launch_bounds__(256) void cvt_w2(const float* __restrict__ src,
                                              unsigned short* __restrict__ dst, size_t total) {
  for (size_t idx = (size_t)blockIdx.x * 256 + threadIdx.x; idx < total;
       idx += (size_t)gridDim.x * 256) {
    int n  = (int)(idx & (IN_DIM - 1));
    int k  = (int)((idx >> 10) & (EX_DIM - 1));
    int ee = (int)(idx >> 22);
    size_t tile = ((size_t)ee * 128 + (k >> 5)) * 64 + (n >> 4);
    dst[tile * 512 + b_tile_idx(k & 31, n & 15)] = f32_bf16_bits(src[idx]);
  }
}

// Wg1[k(1024)][n(4096)] -> tiles [ntile(256)][kstep(32)]
__global__ __launch_bounds__(256) void cvt_wg1(const float* __restrict__ src,
                                               unsigned short* __restrict__ dst, size_t total) {
  for (size_t idx = (size_t)blockIdx.x * 256 + threadIdx.x; idx < total;
       idx += (size_t)gridDim.x * 256) {
    int n = (int)(idx & (EX_DIM - 1));
    int k = (int)(idx >> 12);
    size_t tile = (size_t)(n >> 4) * 32 + (k >> 5);
    dst[tile * 512 + b_tile_idx(k & 31, n & 15)] = f32_bf16_bits(src[idx]);
  }
}

// ---------------- Fused MoE kernel ----------------
// 8 waves = (m 0..3) x (h 0..1). Wave(m,h): GEMM2 accum = rows [m*16,+16) x cols [h*512,+512).

__global__ __launch_bounds__(256)
void moe_fused_kernel(const float* __restrict__ x,
                      const unsigned short* __restrict__ Wg1s,
                      const float* __restrict__ bg1,
                      const float* __restrict__ Wg2, const float* __restrict__ bg2,
                      const unsigned short* __restrict__ W1s, const float* __restrict__ b1,
                      const unsigned short* __restrict__ W2s, const float* __restrict__ b2,
                      float* __restrict__ out) {
  // x block, A-layout: [mtile 4][kstep 32] 1KB blocks = 128 KB
  __shared__ __attribute__((aligned(32))) unsigned short xA[MT * KSTEPS1 * 512];
  // union scratch: gate-phase hf32 (64x128 f32 = 32 KB) / expert-phase hstg (64x128 bf16 = 16 KB)
  __shared__ __attribute__((aligned(32))) unsigned short scratch[16384];
  __shared__ float wg2s[CHUNK * NEXP];     // 4 KB
  __shared__ float logitsL[TM * NEXP];     // 2 KB
  __shared__ float gateL[TM * NEXP];       // 2 KB

  unsigned short* hstg = scratch;          // [mtile 4][kstep 4] 1KB A-layout blocks
  float* hf32 = (float*)scratch;           // [64][128] row-major

  const int tid   = threadIdx.x;
  const int lane  = tid & 31;
  const int wv    = tid >> 5;
  const int m     = wv >> 1;               // M-tile 0..3
  const int h     = wv & 1;                // N-group 0..1
  const int laneN = lane & 15;
  const bool laneHi = lane >= 16;
  const int tok0  = blockIdx.x * TM;

  // ---- Stage x block (64x1024) as bf16 in A-layout ----
  for (int idx = tid; idx < TM * IN_DIM; idx += 256) {
    int token = idx >> 10, k = idx & (IN_DIM - 1);
    float f = x[(size_t)(tok0 + token) * IN_DIM + k];
    xA[((token >> 4) * KSTEPS1 + (k >> 5)) * 512 + a_idx(token & 15, k & 31)] = f32_bf16_bits(f);
  }
  __syncthreads();

  // ---- Gate: softmax(relu(x@Wg1+bg1)@Wg2 + bg2) ----
  float lacc0 = 0.f, lacc1 = 0.f;          // (token=tid>>3, e=tid&7) and token+32
  for (int hc = 0; hc < NCHUNK; ++hc) {
    v8f c1[4];
    v8f zero = {};
#pragma unroll
    for (int j = 0; j < 4; ++j) c1[j] = zero;
#pragma unroll 4
    for (int ks = 0; ks < KSTEPS1; ++ks) {
      v16bf a = load_slot(&xA[(m * KSTEPS1 + ks) * 512 + lane * 16]);
#pragma unroll
      for (int j = 0; j < 4; ++j) {
        const unsigned short* bt = Wg1s + ((size_t)(hc * 8 + h * 4 + j) * 32 + ks) * 512;
        v16bf b = load_slot(bt + lane * 16);
        c1[j] = __builtin_amdgcn_wmma_f32_16x16x32_bf16(false, a, false, b, (short)0, c1[j], false, false);
      }
    }
#pragma unroll
    for (int j = 0; j < 4; ++j) {
      float bias = bg1[(hc * 8 + h * 4 + j) * 16 + laneN];
#pragma unroll
      for (int r = 0; r < 8; ++r) {
        int row = m * 16 + (laneHi ? (r + 8) : r);
        float v = c1[j][r] + bias;
        hf32[row * CHUNK + (h * 4 + j) * 16 + laneN] = v > 0.f ? v : 0.f;
      }
    }
    for (int i = tid; i < CHUNK * NEXP; i += 256)
      wg2s[i] = Wg2[hc * (CHUNK * NEXP) + i];
    __syncthreads();
    {
      int token = tid >> 3, ei = tid & 7;
      float a0 = 0.f, a1 = 0.f;
#pragma unroll 8
      for (int j = 0; j < CHUNK; ++j) {
        float wv2 = wg2s[j * NEXP + ei];
        a0 += hf32[token * CHUNK + j] * wv2;
        a1 += hf32[(token + 32) * CHUNK + j] * wv2;
      }
      lacc0 += a0; lacc1 += a1;
    }
    __syncthreads();
  }
  {
    int token = tid >> 3, ei = tid & 7;
    logitsL[token * NEXP + ei] = lacc0 + bg2[ei];
    logitsL[(token + 32) * NEXP + ei] = lacc1 + bg2[ei];
  }
  __syncthreads();
  if (tid < TM) {
    float mx = -1e30f;
    for (int e = 0; e < NEXP; ++e) mx = fmaxf(mx, logitsL[tid * NEXP + e]);
    float ex[NEXP], s = 0.f;
    for (int e = 0; e < NEXP; ++e) { ex[e] = __expf(logitsL[tid * NEXP + e] - mx); s += ex[e]; }
    float inv = 1.f / s;
    for (int e = 0; e < NEXP; ++e) gateL[tid * NEXP + e] = ex[e] * inv;
  }
  __syncthreads();

  // ---- Experts: out = sum_e gate[:,e] * (relu(x@W1[e]+b1[e]) @ W2[e] + b2[e]) ----
  v8f C[32];
  {
    v8f zero = {};
#pragma unroll
    for (int t = 0; t < 32; ++t) C[t] = zero;
  }

  for (int e = 0; e < NEXP; ++e) {
    float gv[8];
#pragma unroll
    for (int r = 0; r < 8; ++r)
      gv[r] = gateL[(m * 16 + (laneHi ? r + 8 : r)) * NEXP + e];

    for (int hc = 0; hc < NCHUNK; ++hc) {
      // GEMM1: wave computes h tiles (mtile m) x (ntiles h*4..h*4+3)
      v8f c1[4];
      v8f zero = {};
#pragma unroll
      for (int j = 0; j < 4; ++j) c1[j] = zero;
#pragma unroll 4
      for (int ks = 0; ks < KSTEPS1; ++ks) {
        v16bf a = load_slot(&xA[(m * KSTEPS1 + ks) * 512 + lane * 16]);
#pragma unroll
        for (int j = 0; j < 4; ++j) {
          const unsigned short* bt =
              W1s + (((size_t)e * 256 + hc * 8 + h * 4 + j) * 32 + ks) * 512;
          v16bf b = load_slot(bt + lane * 16);
          c1[j] = __builtin_amdgcn_wmma_f32_16x16x32_bf16(false, a, false, b, (short)0, c1[j], false, false);
        }
      }
      // bias + relu + gate-scale -> hstg (A-layout, [mtile][kstep] blocks)
#pragma unroll
      for (int j = 0; j < 4; ++j) {
        float bias = b1[(size_t)e * EX_DIM + (hc * 8 + h * 4 + j) * 16 + laneN];
        int kk = (h * 4 + j) * 16 + laneN;      // 0..127 within chunk
#pragma unroll
        for (int r = 0; r < 8; ++r) {
          int row16 = laneHi ? (r + 8) : r;
          float v = c1[j][r] + bias;
          v = v > 0.f ? v : 0.f;
          v *= gv[r];
          hstg[(m * KSTEPS2 + (kk >> 5)) * 512 + a_idx(row16, kk & 31)] = f32_bf16_bits(v);
        }
      }
      __syncthreads();
      // GEMM2: C += h_chunk(64x128) @ W2[e][hc*128:+128, :]; wave covers cols [h*512,+512)
#pragma unroll
      for (int ks = 0; ks < KSTEPS2; ++ks) {
        v16bf a2 = load_slot(&hstg[(m * KSTEPS2 + ks) * 512 + lane * 16]);
        const unsigned short* bt2 =
            W2s + (((size_t)e * 128 + (hc * 4 + ks)) * 64 + h * 32) * 512;
#pragma unroll
        for (int t = 0; t < 32; ++t) {
          v16bf bb = load_slot(bt2 + (size_t)t * 512 + lane * 16);
          C[t] = __builtin_amdgcn_wmma_f32_16x16x32_bf16(false, a2, false, bb, (short)0, C[t], false, false);
        }
      }
      __syncthreads();
    }
  }

  // ---- Epilogue: add sum_e gate*b2[e], store f32 ----
#pragma unroll
  for (int t = 0; t < 32; ++t) {
    int col = h * 512 + t * 16 + laneN;
    float b2v[NEXP];
#pragma unroll
    for (int e = 0; e < NEXP; ++e) b2v[e] = b2[(size_t)e * IN_DIM + col];
#pragma unroll
    for (int r = 0; r < 8; ++r) {
      int row = m * 16 + (laneHi ? (r + 8) : r);
      float bs = 0.f;
#pragma unroll
      for (int e = 0; e < NEXP; ++e) bs += gateL[row * NEXP + e] * b2v[e];
      out[(size_t)(tok0 + row) * IN_DIM + col] = C[t][r] + bs;
    }
  }
}

extern "C" void kernel_launch(void* const* d_in, const int* in_sizes, int n_in,
                              void* d_out, int out_size, void* d_ws, size_t ws_size,
                              hipStream_t stream) {
  (void)in_sizes; (void)n_in; (void)out_size; (void)ws_size;
  const float* x   = (const float*)d_in[0];
  const float* Wg1 = (const float*)d_in[1];
  const float* bg1 = (const float*)d_in[2];
  const float* Wg2 = (const float*)d_in[3];
  const float* bg2 = (const float*)d_in[4];
  const float* W1  = (const float*)d_in[5];
  const float* b1  = (const float*)d_in[6];
  const float* W2  = (const float*)d_in[7];
  const float* b2  = (const float*)d_in[8];
  float* out = (float*)d_out;

  unsigned short* ws = (unsigned short*)d_ws;
  unsigned short* W1s  = ws + W1S_OFF;
  unsigned short* W2s  = ws + W2S_OFF;
  unsigned short* Wg1s = ws + WG1S_OFF;

  cvt_w1 <<<8192, 256, 0, stream>>>(W1,  W1s,  W1S_ELEMS);
  cvt_w2 <<<8192, 256, 0, stream>>>(W2,  W2s,  W2S_ELEMS);
  cvt_wg1<<<2048, 256, 0, stream>>>(Wg1, Wg1s, WG1S_ELEMS);

  dim3 grid(N_TOKENS / TM);   // 128 workgroups (~1 per WGP)
  dim3 block(256);            // 8 wave32s: 4 M-tiles x 2 N-groups
  moe_fused_kernel<<<grid, block, 0, stream>>>(x, Wg1s, bg1, Wg2, bg2, W1s, b1, W2s, b2, out);
}
